// SimpleGRUSupervisedSeq2Seq_48412871360745
// MI455X (gfx1250) — compile-verified, matrix operand order checked by reading, not run on hardware
//
#include <hip/hip_runtime.h>
#include <hip/hip_bf16.h>

// ---------------- problem constants ----------------
#define B_    64
#define SSRC  128
#define STGT  128
#define E_    512
#define H_    1024
#define H3_   3072
#define V_    8000
#define TD_   127          // decoder steps = STGT-1
#define EH_   1536         // E_ + H_

typedef __attribute__((ext_vector_type(16))) __bf16 v16bf;
typedef __attribute__((ext_vector_type(8)))  __bf16 v8bf;
typedef __attribute__((ext_vector_type(8)))  float  v8f;
typedef __attribute__((ext_vector_type(4)))  unsigned int v4u;

// Load a 16-element bf16 WMMA operand fragment for this lane.
// CDNA5 16-bit A-matrix 16x32 layout: lane L (row = L%16) holds
// k = kbase..kbase+7 (VGPRs 0-3) and k = kbase+16..kbase+23 (VGPRs 4-7),
// where kbase already includes (L/16)*8. Two contiguous 16B loads.
__device__ __forceinline__ v16bf load_frag(const __bf16* __restrict__ p) {
  union { v16bf v; v4u u[2]; } f;
  f.u[0] = *(const v4u*)(p);
  f.u[1] = *(const v4u*)(p + 16);
  return f.v;
}

__device__ __forceinline__ float sigm(float x) { return 1.0f / (1.0f + __expf(-x)); }

// ---------------- WMMA GEMM tile: C[M,N] = A[M,K] @ W[N,K]^T (+bias) ----------
// A, W bf16 row-major (W is the PyTorch [out,in] weight, so both operand
// fragments load identically). One wave -> 16x64 of C (4 N-tiles reuse the A
// fragment). Requires N%64==0, K%32==0, M%16==0 (true for all call sites).
template<bool OUT_BF16>
__device__ __forceinline__ void gemm_tile(const __bf16* __restrict__ A,
                                          const __bf16* __restrict__ W,
                                          const float* __restrict__ bias,
                                          void* __restrict__ Cv,
                                          int N, int K, int lda, long long ldc,
                                          int wid, int lane) {
  const int ngrp  = N >> 6;
  const int mtile = wid / ngrp;
  const int ng    = wid % ngrp;

  const int r     = lane & 15;
  const int khalf = (lane >> 4) << 3;

  const __bf16* Arow = A + (size_t)(mtile * 16 + r) * lda + khalf;
  const __bf16* W0   = W + (size_t)(ng * 64 +  0 + r) * K + khalf;
  const __bf16* W1   = W + (size_t)(ng * 64 + 16 + r) * K + khalf;
  const __bf16* W2   = W + (size_t)(ng * 64 + 32 + r) * K + khalf;
  const __bf16* W3   = W + (size_t)(ng * 64 + 48 + r) * K + khalf;

  v8f acc0 = {}, acc1 = {}, acc2 = {}, acc3 = {};
  for (int k0 = 0; k0 < K; k0 += 32) {
    v16bf a  = load_frag(Arow + k0);
    v16bf b0 = load_frag(W0 + k0);
    v16bf b1 = load_frag(W1 + k0);
    v16bf b2 = load_frag(W2 + k0);
    v16bf b3 = load_frag(W3 + k0);
    acc0 = __builtin_amdgcn_wmma_f32_16x16x32_bf16(false, a, false, b0, (short)0, acc0, false, false);
    acc1 = __builtin_amdgcn_wmma_f32_16x16x32_bf16(false, a, false, b1, (short)0, acc1, false, false);
    acc2 = __builtin_amdgcn_wmma_f32_16x16x32_bf16(false, a, false, b2, (short)0, acc2, false, false);
    acc3 = __builtin_amdgcn_wmma_f32_16x16x32_bf16(false, a, false, b3, (short)0, acc3, false, false);
  }

  const int n0 = ng * 64 + r;
  const float bv0 = bias ? bias[n0]      : 0.f;
  const float bv1 = bias ? bias[n0 + 16] : 0.f;
  const float bv2 = bias ? bias[n0 + 32] : 0.f;
  const float bv3 = bias ? bias[n0 + 48] : 0.f;
  const int mb = mtile * 16 + khalf;           // C layout: m = vgpr + (lane/16)*8
  if constexpr (OUT_BF16) {
    __bf16* C = (__bf16*)Cv;
#pragma unroll
    for (int v = 0; v < 8; ++v) {
      long long ro = (long long)(mb + v) * ldc;
      C[ro + n0]      = (__bf16)(acc0[v] + bv0);
      C[ro + n0 + 16] = (__bf16)(acc1[v] + bv1);
      C[ro + n0 + 32] = (__bf16)(acc2[v] + bv2);
      C[ro + n0 + 48] = (__bf16)(acc3[v] + bv3);
    }
  } else {
    float* C = (float*)Cv;
#pragma unroll
    for (int v = 0; v < 8; ++v) {
      long long ro = (long long)(mb + v) * ldc;
      C[ro + n0]      = acc0[v] + bv0;
      C[ro + n0 + 16] = acc1[v] + bv1;
      C[ro + n0 + 32] = acc2[v] + bv2;
      C[ro + n0 + 48] = acc3[v] + bv3;
    }
  }
}

template<bool OUT_BF16>
__global__ void k_gemm(const __bf16* __restrict__ A, const __bf16* __restrict__ W,
                       const float* __restrict__ bias, void* __restrict__ Cv,
                       int N, int K, int lda, long long ldc, int total_waves) {
  const int wid = blockIdx.x * 8 + (threadIdx.x >> 5);
  if (wid >= total_waves) return;              // wave-uniform: EXEC stays all-1s
  gemm_tile<OUT_BF16>(A, W, bias, Cv, N, K, lda, ldc, wid, threadIdx.x & 31);
}

// Two independent GEMMs in one launch (used for the GRU cell's gi and gh).
// Wave-uniform job selection keeps EXEC all-ones inside each WMMA path.
__global__ void k_gemm2(const __bf16* __restrict__ A0, const __bf16* __restrict__ W0,
                        const float* __restrict__ b0, float* __restrict__ C0,
                        int N0, int K0, int lda0, long long ldc0, int waves0,
                        const __bf16* __restrict__ A1, const __bf16* __restrict__ W1,
                        const float* __restrict__ b1, float* __restrict__ C1,
                        int N1, int K1, int lda1, long long ldc1, int waves1) {
  const int wid  = blockIdx.x * 8 + (threadIdx.x >> 5);
  const int lane = threadIdx.x & 31;
  if (wid < waves0)
    gemm_tile<false>(A0, W0, b0, C0, N0, K0, lda0, ldc0, wid, lane);
  else if (wid < waves0 + waves1)
    gemm_tile<false>(A1, W1, b1, C1, N1, K1, lda1, ldc1, wid - waves0, lane);
}

// ---------------- conversion / setup kernels ----------------
__global__ void k_cvt(const float* __restrict__ s, __bf16* __restrict__ d, int n) {
  int i = blockIdx.x * 256 + threadIdx.x;
  if (i < n) d[i] = (__bf16)s[i];
}

// d[c*R + r] = s[r*C + c]  (transpose fp32 [R,C] -> bf16 [C,R])
__global__ void k_cvt_T(const float* __restrict__ s, __bf16* __restrict__ d, int R, int C) {
  int i = blockIdx.x * 256 + threadIdx.x;
  if (i >= R * C) return;
  int c = i / R, r = i % R;
  d[i] = (__bf16)s[(size_t)r * C + c];
}

// out[t*ostride + e] = emb[toks[t]*E_ + e]   (batched source-embedding gather)
__global__ void k_embed(const int* __restrict__ toks, const float* __restrict__ emb,
                        __bf16* __restrict__ out, int ntok, int ostride) {
  int i = blockIdx.x * 256 + threadIdx.x;
  if (i >= ntok * E_) return;
  int t = i / E_, e = i % E_;
  out[(size_t)t * ostride + e] = (__bf16)emb[(size_t)toks[t] * E_ + e];
}

__global__ void k_init_h(float* __restrict__ h, __bf16* __restrict__ hb) {
  int i = blockIdx.x * 256 + threadIdx.x;
  if (i < B_ * H_) { h[i] = 0.f; hb[i] = (__bf16)0.f; }
}

// ---------------- GRU gate kernels ----------------
__global__ void k_gate_enc(const __bf16* __restrict__ gi_all, const float* __restrict__ gh,
                           float* __restrict__ h, __bf16* __restrict__ hb,
                           __bf16* __restrict__ ehs, int t) {
  int i = blockIdx.x * 256 + threadIdx.x;          // B_*H_ threads
  int b = i >> 10, j = i & 1023;
  const __bf16* gp = gi_all + (size_t)(b * SSRC + t) * H3_;
  const float*  hp = gh + (size_t)b * H3_;
  float ir = (float)gp[j], iz = (float)gp[H_ + j], in_ = (float)gp[2 * H_ + j];
  float hr = hp[j],        hz = hp[H_ + j],        hn  = hp[2 * H_ + j];
  float rr = sigm(ir + hr), zz = sigm(iz + hz);
  float nn = tanhf(in_ + rr * hn);
  float hnew = (1.f - zz) * nn + zz * h[i];
  h[i]  = hnew;
  hb[i] = (__bf16)hnew;
  ehs[(size_t)(b * SSRC + t) * H_ + j] = (__bf16)hnew;
}

__global__ void k_gate_dec(const float* __restrict__ gi, const float* __restrict__ gh,
                           float* __restrict__ h, __bf16* __restrict__ hb) {
  int i = blockIdx.x * 256 + threadIdx.x;
  int b = i >> 10, j = i & 1023;
  const float* gp = gi + (size_t)b * H3_;
  const float* hp = gh + (size_t)b * H3_;
  float ir = gp[j], iz = gp[H_ + j], in_ = gp[2 * H_ + j];
  float hr = hp[j], hz = hp[H_ + j], hn  = hp[2 * H_ + j];
  float rr = sigm(ir + hr), zz = sigm(iz + hz);
  float nn = tanhf(in_ + rr * hn);
  float hnew = (1.f - zz) * nn + zz * h[i];
  h[i]  = hnew;
  hb[i] = (__bf16)hnew;
}

// ---------------- fused attention + context + target-embedding --------------
// One block (256 threads = 8 waves) per batch row b:
//   phase 1: scores[s] = dot(q[b], enc_hs[b,s]) with pad mask (wave32 shuffles)
//   phase 2: softmax over SSRC in LDS
//   phase 3: ctx[h] = sum_s w[s]*enc_hs[b,s,h]  -> x_cat[b, E_:E_+H_]
//   phase 4: x_cat[b, :E_] = dst_emb[out_seq[b, step]]
__global__ void k_attn(const __bf16* __restrict__ q, const __bf16* __restrict__ ehs,
                       const int* __restrict__ src_tok, const int* __restrict__ out_seq,
                       const float* __restrict__ dst_emb, __bf16* __restrict__ x_cat,
                       int step) {
  __shared__ float sc[SSRC];
  __shared__ float red[SSRC];
  __shared__ float wgt[SSRC];
  const int b    = blockIdx.x;
  const int t    = threadIdx.x;
  const int wave = t >> 5;
  const int lane = t & 31;

  // phase 1: 8 waves x 16 s-values each
  const __bf16* qp = q + (size_t)b * H_;
  const __bf16* eb = ehs + (size_t)b * SSRC * H_;
  for (int s = wave; s < SSRC; s += 8) {
    const __bf16* ep = eb + (size_t)s * H_;
    float acc = 0.f;
    for (int c = lane * 8; c < H_; c += 256) {
      v8bf qv = *(const v8bf*)(qp + c);
      v8bf ev = *(const v8bf*)(ep + c);
#pragma unroll
      for (int i = 0; i < 8; ++i) acc += (float)qv[i] * (float)ev[i];
    }
    for (int off = 16; off > 0; off >>= 1) acc += __shfl_down(acc, off, 32);
    if (lane == 0)
      sc[s] = (src_tok[b * SSRC + s] != 0) ? acc : -1e9f;
  }
  __syncthreads();

  // phase 2: softmax over 128 scores (first 128 threads participate in LDS tree)
  float sv = 0.f;
  if (t < SSRC) { sv = sc[t]; red[t] = sv; }
  __syncthreads();
  for (int off = 64; off > 0; off >>= 1) {
    if (t < off) red[t] = fmaxf(red[t], red[t + off]);
    __syncthreads();
  }
  float mx = red[0];
  __syncthreads();
  float w = 0.f;
  if (t < SSRC) { w = __expf(sv - mx); red[t] = w; }
  __syncthreads();
  for (int off = 64; off > 0; off >>= 1) {
    if (t < off) red[t] += red[t + off];
    __syncthreads();
  }
  if (t < SSRC) wgt[t] = w / red[0];
  __syncthreads();

  // phase 3: context -> x_cat[b, E_ + h]
  for (int hh = t; hh < H_; hh += 256) {
    float acc = 0.f;
    for (int s = 0; s < SSRC; ++s) acc += wgt[s] * (float)eb[(size_t)s * H_ + hh];
    x_cat[(size_t)b * EH_ + E_ + hh] = (__bf16)acc;
  }

  // phase 4: target embedding -> x_cat[b, :E_]
  const int tok = out_seq[b * STGT + step];
  const float* ep = dst_emb + (size_t)tok * E_;
  for (int e = t; e < E_; e += 256)
    x_cat[(size_t)b * EH_ + e] = (__bf16)ep[e];
}

// ---------------- final log-softmax (in-place on d_out rows) ----------------
__global__ void k_logsoftmax(float* __restrict__ out) {
  __shared__ float red[256];
  size_t row = (size_t)blockIdx.x * V_;
  int t = threadIdx.x;
  float mx = -3.4e38f;
  for (int v = t; v < V_; v += 256) mx = fmaxf(mx, out[row + v]);
  red[t] = mx; __syncthreads();
  for (int off = 128; off > 0; off >>= 1) {
    if (t < off) red[t] = fmaxf(red[t], red[t + off]);
    __syncthreads();
  }
  mx = red[0]; __syncthreads();
  float sum = 0.f;
  for (int v = t; v < V_; v += 256) sum += __expf(out[row + v] - mx);
  red[t] = sum; __syncthreads();
  for (int off = 128; off > 0; off >>= 1) {
    if (t < off) red[t] += red[t + off];
    __syncthreads();
  }
  float lse = mx + __logf(red[0]);
  __syncthreads();
  for (int v = t; v < V_; v += 256) out[row + v] -= lse;
}

// ---------------- host orchestration ----------------
extern "C" void kernel_launch(void* const* d_in, const int* in_sizes, int n_in,
                              void* d_out, int out_size, void* d_ws, size_t ws_size,
                              hipStream_t stream) {
  const int*   in_seq  = (const int*)d_in[0];
  const int*   out_seq = (const int*)d_in[1];
  const float* src_emb = (const float*)d_in[2];
  const float* enc_Wih = (const float*)d_in[3];
  const float* enc_Whh = (const float*)d_in[4];
  const float* enc_bih = (const float*)d_in[5];
  const float* enc_bhh = (const float*)d_in[6];
  const float* attn_W  = (const float*)d_in[7];
  const float* dst_emb = (const float*)d_in[8];
  const float* dec_Wih = (const float*)d_in[9];
  const float* dec_Whh = (const float*)d_in[10];
  const float* dec_bih = (const float*)d_in[11];
  const float* dec_bhh = (const float*)d_in[12];
  const float* logit_W = (const float*)d_in[13];
  const float* logit_b = (const float*)d_in[14];
  float* out = (float*)d_out;
  (void)in_sizes; (void)n_in; (void)out_size; (void)ws_size;

  char* p = (char*)d_ws;
  auto alloc = [&](size_t bytes) -> void* {
    void* r = (void*)p;
    p += (bytes + 255) & ~(size_t)255;
    return r;
  };
  __bf16* wih_e  = (__bf16*)alloc((size_t)H3_ * E_  * 2);
  __bf16* whh_e  = (__bf16*)alloc((size_t)H3_ * H_  * 2);
  __bf16* wih_d  = (__bf16*)alloc((size_t)H3_ * EH_ * 2);
  __bf16* whh_d  = (__bf16*)alloc((size_t)H3_ * H_  * 2);
  __bf16* awt    = (__bf16*)alloc((size_t)H_  * H_  * 2);   // attn_W^T
  __bf16* lw     = (__bf16*)alloc((size_t)V_  * H_  * 2);
  __bf16* src_e  = (__bf16*)alloc((size_t)B_ * SSRC * E_ * 2);
  __bf16* gi_all = (__bf16*)alloc((size_t)B_ * SSRC * H3_ * 2); // enc input proj (+bih)
  __bf16* ehs    = (__bf16*)alloc((size_t)B_ * SSRC * H_ * 2);  // encoder states
  float*  h      = (float*) alloc((size_t)B_ * H_ * 4);
  __bf16* hb     = (__bf16*)alloc((size_t)B_ * H_ * 2);
  float*  gh     = (float*) alloc((size_t)B_ * H3_ * 4);
  float*  gi_dec = (float*) alloc((size_t)B_ * H3_ * 4);
  __bf16* q_bf   = (__bf16*)alloc((size_t)B_ * H_ * 2);
  __bf16* x_cat  = (__bf16*)alloc((size_t)B_ * EH_ * 2);

  auto cvt = [&](const float* s, __bf16* d, int n) {
    k_cvt<<<(n + 255) / 256, 256, 0, stream>>>(s, d, n);
  };
  // --- one-time (per call) weight conversion to bf16 ---
  cvt(enc_Wih, wih_e, H3_ * E_);
  cvt(enc_Whh, whh_e, H3_ * H_);
  cvt(dec_Wih, wih_d, H3_ * EH_);
  cvt(dec_Whh, whh_d, H3_ * H_);
  cvt(logit_W, lw,    V_  * H_);
  k_cvt_T<<<(H_ * H_ + 255) / 256, 256, 0, stream>>>(attn_W, awt, H_, H_);

  // --- encoder: gather embeddings, batched input projection (8192x512x3072) ---
  k_embed<<<(B_ * SSRC * E_ + 255) / 256, 256, 0, stream>>>(in_seq, src_emb, src_e,
                                                            B_ * SSRC, E_);
  {
    int waves = (B_ * SSRC / 16) * (H3_ / 64);   // 24576
    k_gemm<true><<<waves / 8, 256, 0, stream>>>(src_e, wih_e, enc_bih, gi_all,
                                                H3_, E_, E_, H3_, waves);
  }
  k_init_h<<<B_ * H_ / 256, 256, 0, stream>>>(h, hb);

  // --- encoder recurrence ---
  for (int t = 0; t < SSRC; ++t) {
    int waves = (B_ / 16) * (H3_ / 64);          // 192
    k_gemm<false><<<(waves + 7) / 8, 256, 0, stream>>>(hb, whh_e, enc_bhh, gh,
                                                       H3_, H_, H_, H3_, waves);
    k_gate_enc<<<B_ * H_ / 256, 256, 0, stream>>>(gi_all, gh, h, hb, ehs, t);
  }

  // --- decoder (h restarts at zero) ---
  k_init_h<<<B_ * H_ / 256, 256, 0, stream>>>(h, hb);
  const int wg = (B_ / 16) * (H3_ / 64);         // 192 waves per GRU GEMM
  for (int t = 0; t < TD_; ++t) {
    // q = h @ attn_W  (bf16 out)
    int wq = (B_ / 16) * (H_ / 64);              // 64
    k_gemm<true><<<(wq + 7) / 8, 256, 0, stream>>>(hb, awt, nullptr, q_bf,
                                                   H_, H_, H_, H_, wq);
    // fused: masked scores + softmax + context + target embedding -> x_cat
    k_attn<<<B_, 256, 0, stream>>>(q_bf, ehs, in_seq, out_seq, dst_emb, x_cat, t);
    // GRU cell: gi and gh in a single launch (384 waves)
    k_gemm2<<<(2 * wg) / 8, 256, 0, stream>>>(
        x_cat, wih_d, dec_bih, gi_dec, H3_, EH_, EH_, H3_, wg,
        hb,    whh_d, dec_bhh, gh,     H3_, H_,  H_,  H3_, wg);
    k_gate_dec<<<B_ * H_ / 256, 256, 0, stream>>>(gi_dec, gh, h, hb);
    // logits -> d_out[b, t, :] with row stride TD_*V_
    int wl = (B_ / 16) * (V_ / 64);              // 500
    k_gemm<false><<<(wl + 7) / 8, 256, 0, stream>>>(hb, lw, logit_b,
                                                    out + (size_t)t * V_,
                                                    V_, H_, H_,
                                                    (long long)TD_ * V_, wl);
  }

  // --- final log_softmax over vocab, in place ---
  k_logsoftmax<<<B_ * TD_, 256, 0, stream>>>(out);
}
